// ResidualGatedAssociativeLM_34368328303133
// MI455X (gfx1250) — matrix operand order repaired and checked by reading, not compiled, not run on hardware
//
#include <hip/hip_runtime.h>
#include <math.h>

typedef __attribute__((ext_vector_type(16))) _Float16 v16h;
typedef __attribute__((ext_vector_type(8)))  float    v8f;

#define DEV_INLINE __device__ __forceinline__

static constexpr int Bb = 2, Ss = 1024, Ee = 512, Hh = 1024, Mm = 256, Vv = 32000;
static constexpr int BS = Bb * Ss;

// ---------------- CDNA5 async global->LDS (ASYNCcnt-tracked) ----------------
// VDST operand = u32 LDS byte address. Flat pointers to LDS carry the LDS offset
// in addr[31:0] (aperture rule), so truncation yields the correct VDST value.
DEV_INLINE void async_copy_b128(const void* g, void* l) {
  unsigned lds_off = (unsigned)(uintptr_t)l;
  unsigned long long ga = (unsigned long long)(uintptr_t)g;
  asm volatile("global_load_async_to_lds_b128 %0, %1, off"
               :: "v"(lds_off), "v"(ga) : "memory");
}
DEV_INLINE void wait_async0() {
  asm volatile("s_wait_asynccnt 0x0" ::: "memory");
}

// ---------------- reductions (wave32) ----------------
DEV_INLINE float warpSum(float v) {
#pragma unroll
  for (int o = 16; o > 0; o >>= 1) v += __shfl_down(v, o, 32);
  return v;
}
DEV_INLINE float warpMax(float v) {
#pragma unroll
  for (int o = 16; o > 0; o >>= 1) v = fmaxf(v, __shfl_down(v, o, 32));
  return v;
}
DEV_INLINE float blockSum(float v, float* red) {  // 256 threads, red[8]
  int w = threadIdx.x >> 5, l = threadIdx.x & 31;
  v = warpSum(v);
  if (l == 0) red[w] = v;
  __syncthreads();
  float t = (threadIdx.x < 8) ? red[threadIdx.x] : 0.f;
  if (w == 0) { t = warpSum(t); if (l == 0) red[0] = t; }
  __syncthreads();
  float r = red[0];
  __syncthreads();
  return r;
}
DEV_INLINE float blockMax(float v, float* red) {
  int w = threadIdx.x >> 5, l = threadIdx.x & 31;
  v = warpMax(v);
  if (l == 0) red[w] = v;
  __syncthreads();
  float t = (threadIdx.x < 8) ? red[threadIdx.x] : -INFINITY;
  if (w == 0) { t = warpMax(t); if (l == 0) red[0] = t; }
  __syncthreads();
  float r = red[0];
  __syncthreads();
  return r;
}

// ---------------- embedding gather ----------------
__global__ __launch_bounds__(128)
void gather_emb(const int* __restrict__ ids, const float* __restrict__ emb_W,
                float* __restrict__ emb) {
  int t = blockIdx.x;
  int id = ids[t];
  const float4* src = (const float4*)(emb_W + (size_t)id * Ee);
  float4* dst = (float4*)(emb + (size_t)t * Ee);
  for (int e = threadIdx.x; e < Ee / 4; e += 128) dst[e] = src[e];
}

// ---------------- WMMA GEMM: C = (A @ W^T + bias) * scale ----------------
// A: [MM,K] f32, W: [NN,K] f32, C: [MM,NN] f32. MM%64==0, NN%128==0, K%32==0.
// Block tile 64x128: 8 waves, each wave = 4 stacked 16x16 WMMA tiles sharing one
// B fragment. K staged in 32-wide chunks via double-buffered async global->LDS.
__global__ __launch_bounds__(256)
void gemm_wmma(const float* __restrict__ A, const float* __restrict__ W,
               const float* __restrict__ bias, const float* __restrict__ scale,
               float* __restrict__ C, int MM, int NN, int K, int nt_store) {
  constexpr int TK = 32;
  constexpr int RP = TK + 4;  // padded row stride (floats): 144B rows, 16B aligned
  __shared__ float Abuf[2][64 * RP];
  __shared__ float Wbuf[2][128 * RP];
  const int m0 = blockIdx.x * 64, n0 = blockIdx.y * 128;
  const int tid = threadIdx.x, wave = tid >> 5, lane = tid & 31;
  const int hf = lane >> 4, l16 = lane & 15;

  const v8f zero8 = {0.f, 0.f, 0.f, 0.f, 0.f, 0.f, 0.f, 0.f};
  v8f acc[4] = {zero8, zero8, zero8, zero8};

  auto stage = [&](int buf, int k0) {
    // A chunk 64x32: 512 16B quads; W chunk 128x32: 1024 quads. 6 asyncs/thread.
    for (int q = tid; q < 512; q += 256) {
      int r = q >> 3, c4 = (q & 7) * 4;
      async_copy_b128(&A[(size_t)(m0 + r) * K + k0 + c4], &Abuf[buf][r * RP + c4]);
    }
    for (int q = tid; q < 1024; q += 256) {
      int r = q >> 3, c4 = (q & 7) * 4;
      async_copy_b128(&W[(size_t)(n0 + r) * K + k0 + c4], &Wbuf[buf][r * RP + c4]);
    }
  };

  stage(0, 0);
  int buf = 0;
  for (int k0 = 0; k0 < K; k0 += TK) {
    wait_async0();     // my staged copies for `buf` are done
    __syncthreads();   // everyone's copies done; prev chunk fully consumed
    if (k0 + TK < K) stage(buf ^ 1, k0 + TK);  // overlap next chunk with compute

    // B fragment (32x16): k = e + 16*half, n = lane&15 -> 4x ds_load_b128 + cvt
    v16h bf;
    const float* wrow = &Wbuf[buf][(wave * 16 + l16) * RP + 16 * hf];
#pragma unroll
    for (int e = 0; e < 16; e++) bf[e] = (_Float16)wrow[e];

#pragma unroll
    for (int mt = 0; mt < 4; mt++) {
      v16h af;  // A fragment (16x32): k = 16*half + e, same order as B
      const float* arow = &Abuf[buf][(mt * 16 + l16) * RP];
#pragma unroll
      for (int e = 0; e < 16; e++) {
        int ka = 16 * hf + e;
        af[e] = (_Float16)arow[ka];
      }
      acc[mt] = __builtin_amdgcn_wmma_f32_16x16x32_f16(false, af, false, bf,
                                                       (short)0, acc[mt], false, false);
    }
    buf ^= 1;
  }

  const int col = n0 + wave * 16 + l16;
  const float bv = bias ? bias[col] : 0.f;
  const float sc = scale ? scale[0] : 1.f;
#pragma unroll
  for (int mt = 0; mt < 4; mt++) {
#pragma unroll
    for (int r = 0; r < 8; r++) {  // C/D: row = r + 8*half, col = lane&15
      int row = m0 + mt * 16 + r + 8 * hf;
      float v = (acc[mt][r] + bv) * sc;
      float* p = &C[(size_t)row * NN + col];
      if (nt_store) __builtin_nontemporal_store(v, p);  // logits: straight to HBM
      else *p = v;
    }
  }
}

// ---------------- GRU: cooperative persistent scan ----------------
__global__ void gru_init(float* hbuf, unsigned* bar) {
  int t = blockIdx.x * blockDim.x + threadIdx.x;
  if (t < 2 * Bb * Hh) hbuf[t] = 0.f;
  if (t < 2) bar[t] = 0u;
}

DEV_INLINE void gridBarrier(unsigned* cnt, volatile unsigned* gen, unsigned nwg) {
  __syncthreads();
  if (threadIdx.x == 0) {
    unsigned my = *gen;
    __threadfence();
    if (atomicAdd(cnt, 1u) == nwg - 1u) {
      *cnt = 0u;
      __threadfence();
      atomicAdd((unsigned*)gen, 1u);
    } else {
      while (*gen == my) { __builtin_amdgcn_s_sleep(2); }
    }
  }
  __syncthreads();
  __threadfence();
}

// 64 WGs; WG owns 16 hidden dims => 48 rows of W_hh cached in LDS (192KB of 320KB).
__global__ __launch_bounds__(256)
void gru_scan(const float* __restrict__ x_proj,  // [B,S,3H] (r,z,n)
              const float* __restrict__ W_hh,    // [3H,H]
              const float* __restrict__ b_hh,    // [3H]
              float* __restrict__ states,        // [B,S,H]
              float* __restrict__ hbuf,          // [2][B][H] ping-pong
              unsigned* __restrict__ bar) {
  constexpr int HS = 16, ROWS = 48, NWG = 64;
  extern __shared__ float sm[];
  float* Wl = sm;              // [ROWS][H]
  float* hl = Wl + ROWS * Hh;  // [2][H]
  float* hg = hl + 2 * Hh;     // [2][ROWS]
  float* bl = hg + 2 * ROWS;   // [ROWS]
  const int blk = blockIdx.x, tid = threadIdx.x, wave = tid >> 5, lane = tid & 31;
  const int j0 = blk * HS;

  // async-stage this WG's W_hh slice (gate g in {0,1,2} x 16 hidden rows)
  for (int q = tid; q < ROWS * (Hh / 4); q += 256) {
    int rr = q / (Hh / 4), c4 = (q % (Hh / 4)) * 4;
    int gg = rr >> 4, jj = rr & 15;
    async_copy_b128(&W_hh[((size_t)(gg * Hh + j0 + jj)) * Hh + c4], &Wl[rr * Hh + c4]);
  }
  if (tid < ROWS) { int gg = tid >> 4, jj = tid & 15; bl[tid] = b_hh[gg * Hh + j0 + jj]; }
  wait_async0();

  int cur = 0;
  for (int t = 0; t < Ss; ++t) {
    __syncthreads();  // all waves done reading hl of previous step
    for (int q = tid; q < 2 * Hh / 4; q += 256)  // broadcast h_prev (8KB) via async
      async_copy_b128(&hbuf[cur * 2 * Hh + q * 4], &hl[q * 4]);
    wait_async0();
    __syncthreads();
    // 8 waves x 6 rows: hg = h_prev @ W_slice^T for both batches
#pragma unroll
    for (int r6 = 0; r6 < 6; r6++) {
      int rr = wave * 6 + r6;
      const float* wr = &Wl[rr * Hh];
      float s0 = 0.f, s1 = 0.f;
      for (int c = lane; c < Hh; c += 32) {
        float wv = wr[c];
        s0 += wv * hl[c];
        s1 += wv * hl[Hh + c];
      }
      s0 = warpSum(s0); s1 = warpSum(s1);
      if (lane == 0) { hg[rr] = s0 + bl[rr]; hg[ROWS + rr] = s1 + bl[rr]; }
    }
    __syncthreads();
    if (tid < 32) {  // 2 batches x 16 hidden
      int bb = tid >> 4, jj = tid & 15, j = j0 + jj;
      size_t xb = ((size_t)bb * Ss + t) * (3 * Hh);
      float xr = x_proj[xb + j];
      float xz = x_proj[xb + Hh + j];
      float xn = x_proj[xb + 2 * Hh + j];
      float hr = hg[bb * ROWS + jj];
      float hz = hg[bb * ROWS + 16 + jj];
      float hn = hg[bb * ROWS + 32 + jj];
      float r = 1.f / (1.f + __expf(-(xr + hr)));
      float z = 1.f / (1.f + __expf(-(xz + hz)));
      float n = tanhf(xn + r * hn);
      float hp = hl[bb * Hh + j];
      float hnew = (1.f - z) * n + z * hp;
      states[((size_t)bb * Ss + t) * Hh + j] = hnew;
      hbuf[(cur ^ 1) * 2 * Hh + bb * Hh + j] = hnew;
    }
    __threadfence();
    gridBarrier(bar, bar + 1, NWG);
    cur ^= 1;
  }
}

// ---------------- mixed = rmsnorm(resid_lambda*states + inres) ----------------
__global__ __launch_bounds__(256)
void mixed_rms(const float* __restrict__ states, const float* __restrict__ inres,
               const float* __restrict__ resid_lambda, float* __restrict__ mixed) {
  __shared__ float red[8];
  const int t = blockIdx.x, tid = threadIdx.x;
  const float rl = resid_lambda[0];
  const size_t base = (size_t)t * Hh;
  float v[4];
  float ss = 0.f;
#pragma unroll
  for (int i = 0; i < 4; i++) {
    int c = tid + i * 256;
    float x = rl * states[base + c] + inres[base + c];
    v[i] = x;
    ss += x * x;
  }
  float tot = blockSum(ss, red);
  float inv = rsqrtf(tot * (1.0f / Hh) + 1.1920929e-07f);
#pragma unroll
  for (int i = 0; i < 4; i++) mixed[base + tid + i * 256] = v[i] * inv;
}

// ---------------- strictly-causal attention + gated vocab scatter ----------------
// One block per (b,i). Deterministic scatter: thread t exclusively owns ids with id%256==t.
__global__ __launch_bounds__(256)
void attn_scatter(const float* __restrict__ qm, const float* __restrict__ km,
                  const float* __restrict__ mixed, const float* __restrict__ gate_W,
                  const float* __restrict__ gate_b, const float* __restrict__ mem_scale,
                  const int* __restrict__ ids, float* __restrict__ out) {
  const int i = blockIdx.x, b = blockIdx.y;
  if (i == 0) return;  // strictly-causal row 0 contributes nothing
  __shared__ float qs[Mm];
  __shared__ float sc[Ss];
  __shared__ int idl[Ss];
  __shared__ float red[8];
  const int tid = threadIdx.x, wave = tid >> 5, lane = tid & 31;
  const size_t rowq = (size_t)b * Ss + i;

  qs[tid] = qm[rowq * Mm + tid];
  for (int e = tid; e < Ss; e += 256) idl[e] = ids[b * Ss + e];

  const float* mrow = &mixed[rowq * Hh];
  float g = 0.f;
  for (int c = tid; c < Hh; c += 256) g += mrow[c] * gate_W[c];
  float gt = blockSum(g, red);  // syncs also publish qs/idl
  float gscale = (1.f / (1.f + __expf(-(gt + gate_b[0])))) * mem_scale[0];

  // scores for j < i: wave-cooperative 256-dim dots (coalesced over d)
  for (int j = wave; j < i; j += 8) {
    const float* krow = &km[((size_t)b * Ss + j) * Mm];
    float s = 0.f;
#pragma unroll
    for (int d = lane; d < Mm; d += 32) s += qs[d] * krow[d];
    s = warpSum(s);
    if (lane == 0) sc[j] = s * 0.0625f;  // 1/sqrt(256)
  }
  __syncthreads();

  float mloc = -INFINITY;
  for (int j = tid; j < i; j += 256) mloc = fmaxf(mloc, sc[j]);
  float mx = blockMax(mloc, red);
  float sloc = 0.f;
  for (int j = tid; j < i; j += 256) {
    float e = __expf(sc[j] - mx);
    sc[j] = e;
    sloc += e;
  }
  float ssum = blockSum(sloc, red);
  float fac = gscale / ssum;

  float* orow = out + rowq * (size_t)Vv;
  for (int j = 0; j < i; j++) {
    int id = idl[j];
    if ((id & 255) == tid) orow[id] += sc[j] * fac;  // unique owner -> no races
  }
}

// ---------------- host-side launch ----------------
extern "C" void kernel_launch(void* const* d_in, const int* in_sizes, int n_in,
                              void* d_out, int out_size, void* d_ws, size_t ws_size,
                              hipStream_t stream) {
  (void)in_sizes; (void)n_in; (void)out_size; (void)ws_size;
  const int*   ids          = (const int*)d_in[0];
  const float* emb_W        = (const float*)d_in[1];
  const float* W_ih         = (const float*)d_in[2];
  const float* W_hh         = (const float*)d_in[3];
  const float* b_ih         = (const float*)d_in[4];
  const float* b_hh         = (const float*)d_in[5];
  const float* in_res_W     = (const float*)d_in[6];
  const float* q_W          = (const float*)d_in[7];
  const float* q_b          = (const float*)d_in[8];
  const float* k_W          = (const float*)d_in[9];
  const float* k_b          = (const float*)d_in[10];
  const float* gate_W       = (const float*)d_in[11];
  const float* gate_b       = (const float*)d_in[12];
  const float* h2e_W        = (const float*)d_in[13];
  const float* h2e_b        = (const float*)d_in[14];
  const float* out_bias     = (const float*)d_in[15];
  const float* memory_scale = (const float*)d_in[16];
  const float* resid_lambda = (const float*)d_in[17];
  const float* input_lambda = (const float*)d_in[18];
  float* out = (float*)d_out;

  // workspace layout (with reuse): total ~54.5 MB
  float* ws     = (float*)d_ws;
  float* emb    = ws;                        // BS*E   = 1,048,576 f
  float* xproj  = emb + (size_t)BS * Ee;     // BS*3H  = 6,291,456 f
  float* inres  = xproj + (size_t)BS * 3 * Hh;
  float* states = inres + (size_t)BS * Hh;
  float* mixed  = states + (size_t)BS * Hh;
  float* hbuf   = mixed + (size_t)BS * Hh;   // 4096 f
  unsigned* bar = (unsigned*)(hbuf + 2 * Bb * Hh);
  float* hide   = emb;                       // reuse: emb dead after in_res GEMM
  float* qbuf   = xproj;                     // reuse: xproj dead after GRU
  float* kbuf   = xproj + (size_t)BS * Mm;

  constexpr size_t gru_lds = (size_t)(48 * Hh + 2 * Hh + 2 * 48 + 48) * sizeof(float);

  gather_emb<<<BS, 128, 0, stream>>>(ids, emb_W, emb);
  gemm_wmma<<<dim3(BS / 64, 3 * Hh / 128), 256, 0, stream>>>(
      emb, W_ih, b_ih, nullptr, xproj, BS, 3 * Hh, Ee, 0);
  gemm_wmma<<<dim3(BS / 64, Hh / 128), 256, 0, stream>>>(
      emb, in_res_W, nullptr, input_lambda, inres, BS, Hh, Ee, 0);
  gru_init<<<16, 256, 0, stream>>>(hbuf, bar);
  gru_scan<<<64, 256, gru_lds, stream>>>(xproj, W_hh, b_hh, states, hbuf, bar);
  mixed_rms<<<BS, 256, 0, stream>>>(states, inres, resid_lambda, mixed);
  gemm_wmma<<<dim3(BS / 64, Ee / 128), 256, 0, stream>>>(
      mixed, h2e_W, h2e_b, nullptr, hide, BS, Ee, Hh, 0);
  gemm_wmma<<<dim3(BS / 64, Vv / 128), 256, 0, stream>>>(
      hide, emb_W, out_bias, nullptr, out, BS, Vv, Ee, 1);  // NT stores to HBM
  gemm_wmma<<<dim3(BS / 64, Mm / 128), 256, 0, stream>>>(
      mixed, q_W, q_b, nullptr, qbuf, BS, Mm, Hh, 0);
  gemm_wmma<<<dim3(BS / 64, Mm / 128), 256, 0, stream>>>(
      mixed, k_W, k_b, nullptr, kbuf, BS, Mm, Hh, 0);
  attn_scatter<<<dim3(Ss, Bb), 256, 0, stream>>>(
      qbuf, kbuf, mixed, gate_W, gate_b, memory_scale, ids, out);
}
